// LSTM_Vanilla_76519137345618
// MI455X (gfx1250) — compile-verified, matrix operand order checked by reading, not run on hardware
//
#include <hip/hip_runtime.h>
#include <hip/hip_bf16.h>
#include <stdint.h>

// Problem sizes (fixed by the reference)
#define B_DIM 8192
#define I_DIM 1024
#define H_DIM 1024
#define K_DIM (I_DIM + H_DIM)   // 2048 : fused K = [x | h_prev]
#define BM 128                  // block M tile (8 waves x 16 rows)
#define BN 32                   // block N tile per gate (2 x 16)
#define BK 32                   // K step per WMMA
#define KT (K_DIM / BK)         // 64 K-iterations
#define WTILE_BYTES 0x2000u     // one W LDS buffer: 4*32*32*2 = 8 KB

typedef __attribute__((ext_vector_type(16))) __bf16 v16bf;
typedef __attribute__((ext_vector_type(8)))  __bf16 v8bf;
typedef __attribute__((ext_vector_type(8)))  float  v8f;

__device__ __forceinline__ uint16_t f32_to_bf16_rne(float f) {
    uint32_t u = __float_as_uint(f);
    uint32_t r = u + 0x7FFFu + ((u >> 16) & 1u);
    return (uint16_t)(r >> 16);
}

__device__ __forceinline__ float fast_sigmoid(float x) {
    return 1.0f / (1.0f + __expf(-x));
}
__device__ __forceinline__ float fast_tanh(float x) {
    // tanh(x) = sign(x) * (1 - e^{-2|x|}) / (1 + e^{-2|x|}); overflow-safe.
    float t = __expf(-2.0f * fabsf(x));
    float r = (1.0f - t) / (1.0f + t);
    return copysignf(r, x);
}

// ---------------------------------------------------------------------------
// Pack A = [x | h_prev]  ->  bf16 row-major [B_DIM, K_DIM]
// ---------------------------------------------------------------------------
__global__ __launch_bounds__(256)
void pack_A_kernel(const float* __restrict__ x, const float* __restrict__ h,
                   uint16_t* __restrict__ A) {
    int idx = blockIdx.x * 256 + threadIdx.x;       // over B_DIM * K_DIM
    int b = idx >> 11;                              // K_DIM == 2048
    int k = idx & (K_DIM - 1);
    float v = (k < I_DIM) ? x[(size_t)b * I_DIM + k]
                          : h[(size_t)b * H_DIM + (k - I_DIM)];
    A[idx] = f32_to_bf16_rne(v);
}

// ---------------------------------------------------------------------------
// Pack W = [Wx | Wh]  ->  bf16 row-major [4*H_DIM, K_DIM]   (row = gate*H + n)
// ---------------------------------------------------------------------------
__global__ __launch_bounds__(256)
void pack_W_kernel(const float* __restrict__ Wx, const float* __restrict__ Wh,
                   uint16_t* __restrict__ W) {
    int idx = blockIdx.x * 256 + threadIdx.x;       // over 4 * H_DIM * K_DIM
    int k  = idx & (K_DIM - 1);
    int n  = (idx >> 11) & (H_DIM - 1);
    int g  = idx >> 21;
    float v = (k < I_DIM)
        ? Wx[((size_t)g * H_DIM + n) * I_DIM + k]
        : Wh[((size_t)g * H_DIM + n) * H_DIM + (k - I_DIM)];
    W[idx] = f32_to_bf16_rne(v);
}

// ---------------------------------------------------------------------------
// Fused LSTM GEMM + epilogue.
//   gates[b, g, n] = sum_k A[b,k] * W[g*H+n, k] + bh[g,n]
// Block: 256 threads = 8 wave32. Tile: BM=128 rows, BN=32 cols, all 4 gates.
// W tile double-buffered in LDS (runtime bufOff so the K loop stays a single
// body and the WMMA C->D operands coalesce in place), filled by
// GLOBAL_LOAD_ASYNC_TO_LDS_B128 (ASYNCcnt path); A frags from global (L2).
// ---------------------------------------------------------------------------
__global__ __launch_bounds__(256)
void lstm_gemm_kernel(const uint16_t* __restrict__ A,      // [B_DIM, K_DIM] bf16
                      const uint16_t* __restrict__ W,      // [4*H_DIM, K_DIM] bf16
                      const float*    __restrict__ bh,     // [4, H_DIM]
                      const float*    __restrict__ c_prev, // [B_DIM, H_DIM]
                      float*          __restrict__ h_next, // [B_DIM, H_DIM]
                      float*          __restrict__ c_next) // [B_DIM, H_DIM]
{
    __shared__ __align__(16) uint16_t Wlds[2][4][BN][BK];  // 16 KB, double buffered

    const int tid  = threadIdx.x;
    const int wave = tid >> 5;
    const int lane = tid & 31;
    const int half = lane >> 4;      // 0: lanes 0-15, 1: lanes 16-31
    const int l16  = lane & 15;

    const int row_base = blockIdx.x * BM + wave * 16;   // batch rows of this wave
    const int col_base = blockIdx.y * BN;               // H columns of this block

    v8f acc[4][2];
    #pragma unroll
    for (int g = 0; g < 4; ++g)
        #pragma unroll
        for (int ns = 0; ns < 2; ++ns)
            #pragma unroll
            for (int e = 0; e < 8; ++e)
                acc[g][ns][e] = 0.0f;

    // The 8 KB W tile is linear in chunk id c (c*16 bytes): chunk c holds
    // row r = c>>2 (= g*32+nLocal), 16B piece j = c&3.  Each thread fills
    // chunks {tid, tid+256} with async global->LDS B128 transfers.
    const uint32_t ldsBase32 = (uint32_t)(uintptr_t)(&Wlds[0][0][0][0]);
    auto issueW = [&](uint32_t bufOff, int k0) {
        #pragma unroll
        for (int i = 0; i < 2; ++i) {
            int c  = tid + i * 256;           // 0..511
            int j  = c & 3;                   // 16B chunk within row
            int r  = c >> 2;                  // 0..127 = g*32 + nLocal
            int g  = r >> 5;
            int nl = r & 31;
            const uint16_t* src =
                W + ((size_t)g * H_DIM + col_base + nl) * K_DIM + k0 + j * 8;
            uint32_t lds_off = ldsBase32 + bufOff + (uint32_t)c * 16u;
            uint64_t gaddr   = (uint64_t)(uintptr_t)src;
            asm volatile("global_load_async_to_lds_b128 %0, %1, off"
                         :: "v"(lds_off), "v"(gaddr) : "memory");
        }
    };

    issueW(0, 0);
    uint32_t bufOff = 0;

    // Per-lane A base: row = row_base + l16 (same for both lane halves);
    // K chunks at {k0 + 8*half} and {k0 + 16 + 8*half} per the ISA A-layout.
    const uint16_t* aBase = A + (size_t)(row_base + l16) * K_DIM + 8 * half;
    // Per-lane LDS base for B fragments (column l16, K-chunk by lane half).
    const char* ldsLane =
        (const char*)&Wlds[0][0][0][0] + ((size_t)l16 * BK + 8 * half) * 2;

    #pragma unroll 1
    for (int kt = 0; kt < KT; ++kt) {
        // Publish tile[bufOff]: wait for this wave's async fills, then barrier.
        asm volatile("s_wait_asynccnt 0x0" ::: "memory");
        __syncthreads();

        if (kt + 1 < KT) issueW(bufOff ^ WTILE_BYTES, (kt + 1) * BK);

        const int k0 = kt * BK;

        // A fragment (16 rows x 32 K, bf16): two 16B chunks per lane.
        v8bf a0 = *(const v8bf*)(aBase + k0);
        v8bf a1 = *(const v8bf*)(aBase + k0 + 16);
        v16bf afrag = __builtin_shufflevector(a0, a1,
            0,1,2,3,4,5,6,7,8,9,10,11,12,13,14,15);

        // 8 B fragments from LDS (immediate ds offsets off one address VGPR),
        // then 8 independent back-to-back WMMAs accumulating in place.
        const char* bBase = ldsLane + bufOff;
        #pragma unroll
        for (int g = 0; g < 4; ++g) {
            #pragma unroll
            for (int ns = 0; ns < 2; ++ns) {
                const uint16_t* bp =
                    (const uint16_t*)(bBase + (size_t)(g * 2048 + ns * 1024));
                v8bf b0 = *(const v8bf*)bp;
                v8bf b1 = *(const v8bf*)(bp + 16);
                v16bf bfrag = __builtin_shufflevector(b0, b1,
                    0,1,2,3,4,5,6,7,8,9,10,11,12,13,14,15);
                acc[g][ns] = __builtin_amdgcn_wmma_f32_16x16x32_bf16(
                    false, afrag, false, bfrag, (short)0, acc[g][ns],
                    false, false);
            }
        }
        bufOff ^= WTILE_BYTES;
    }

    // Fused LSTM epilogue on the accumulator fragments.
    // C/D layout: element (vgpr v, lane l) -> row = v + 8*(l>=16), col = l%16.
    #pragma unroll
    for (int ns = 0; ns < 2; ++ns) {
        const int gcol = col_base + ns * 16 + l16;
        const float bi = bh[0 * H_DIM + gcol];
        const float bf = bh[1 * H_DIM + gcol];
        const float bg = bh[2 * H_DIM + gcol];
        const float bo = bh[3 * H_DIM + gcol];
        #pragma unroll
        for (int v = 0; v < 8; ++v) {
            const int grow = row_base + 8 * half + v;
            const size_t off = (size_t)grow * H_DIM + gcol;
            const float iv = fast_sigmoid(acc[0][ns][v] + bi);
            const float fv = fast_sigmoid(acc[1][ns][v] + bf);
            const float gv = fast_tanh   (acc[2][ns][v] + bg);
            const float ov = fast_sigmoid(acc[3][ns][v] + bo);
            const float cn = fv * c_prev[off] + iv * gv;
            const float hn = ov * fast_tanh(cn);
            h_next[off] = hn;
            c_next[off] = cn;
        }
    }
}

// ---------------------------------------------------------------------------
// Host launcher
// ---------------------------------------------------------------------------
extern "C" void kernel_launch(void* const* d_in, const int* in_sizes, int n_in,
                              void* d_out, int out_size, void* d_ws, size_t ws_size,
                              hipStream_t stream) {
    const float* x   = (const float*)d_in[0];   // [B, I]
    const float* hp  = (const float*)d_in[1];   // [B, H]
    const float* cp  = (const float*)d_in[2];   // [B, H]
    const float* Wx  = (const float*)d_in[3];   // [4, H, I]
    const float* Wh  = (const float*)d_in[4];   // [4, H, H]
    const float* bh  = (const float*)d_in[5];   // [4, H]

    float* h_next = (float*)d_out;                            // first output
    float* c_next = (float*)d_out + (size_t)B_DIM * H_DIM;    // second output

    // Workspace: A_bf16 [B, 2048] (32 MB) then W_bf16 [4096, 2048] (16 MB)
    uint16_t* Abf = (uint16_t*)d_ws;
    uint16_t* Wbf = Abf + (size_t)B_DIM * K_DIM;

    pack_A_kernel<<<(B_DIM * K_DIM) / 256, 256, 0, stream>>>(x, hp, Abf);
    pack_W_kernel<<<(4 * H_DIM * K_DIM) / 256, 256, 0, stream>>>(Wx, Wh, Wbf);

    dim3 grid(B_DIM / BM, H_DIM / BN);   // 64 x 32 = 2048 workgroups
    lstm_gemm_kernel<<<grid, 256, 0, stream>>>(Abf, Wbf, bh, cp, h_next, c_next);
}
